// Block_2860448219398
// MI455X (gfx1250) — matrix operand order repaired and checked by reading, not compiled, not run on hardware
//
#include <hip/hip_runtime.h>
#include <stdint.h>

// ---------------------------------------------------------------------------
// Types for WMMA fragments (wave32, gfx1250)
// ---------------------------------------------------------------------------
typedef _Float16 h8v  __attribute__((ext_vector_type(8)));
typedef _Float16 v16h __attribute__((ext_vector_type(16)));
typedef float    v8f  __attribute__((ext_vector_type(8)));

union H16 { v16h v; h8v h[2]; };

__device__ __forceinline__ v8f wmma_f16(v16h a, v16h b, v8f c) {
    return __builtin_amdgcn_wmma_f32_16x16x32_f16(
        /*neg_a=*/false, a, /*neg_b=*/false, b,
        /*c_mod=*/(short)0, c, /*reuse_a=*/false, /*reuse_b=*/false);
}

// CDNA5 async copy: one wave-wide instruction moves 16B/lane global->LDS,
// tracked by ASYNCcnt (no VGPR round trip). Inline asm = portable across
// ROCm7.2 / amdgpu-toolchain builtin arity differences.
__device__ __forceinline__ void async_ld_b128(unsigned lds_addr, const void* g) {
    asm volatile("global_load_async_to_lds_b128 %0, %1, off"
                 :: "v"(lds_addr), "v"((unsigned long long)(uintptr_t)g)
                 : "memory");
}
#define WAIT_ASYNC_LE_2() asm volatile("s_wait_asynccnt 2" ::: "memory")
#define WAIT_ASYNC_0()    asm volatile("s_wait_asynccnt 0" ::: "memory")

#define CC   1024
#define TT   2048
#define BB   2
#define HH   16

// ---------------------------------------------------------------------------
// Fused LayerNorm: row of C=1024, 256 threads/block (8 waves), outputs f32+f16
// ---------------------------------------------------------------------------
__global__ __launch_bounds__(256) void ln_kernel(
    const float* __restrict__ x, const float* __restrict__ g,
    const float* __restrict__ be, float* __restrict__ xo,
    _Float16* __restrict__ xh)
{
    const int row = blockIdx.x;
    const int tid = threadIdx.x;
    const float* xr = x + (size_t)row * CC;

    float4 v = ((const float4*)xr)[tid];
    float s  = v.x + v.y + v.z + v.w;
    float ss = v.x * v.x + v.y * v.y + v.z * v.z + v.w * v.w;
    for (int m = 1; m < 32; m <<= 1) {
        s  += __shfl_xor(s, m);
        ss += __shfl_xor(ss, m);
    }
    __shared__ float red[16];
    if ((tid & 31) == 0) { red[tid >> 5] = s; red[8 + (tid >> 5)] = ss; }
    __syncthreads();
    float sum = 0.f, sq = 0.f;
    for (int i = 0; i < 8; i++) { sum += red[i]; sq += red[8 + i]; }
    const float mu = sum * (1.0f / CC);
    const float var = sq * (1.0f / CC) - mu * mu;
    const float rs = rsqrtf(var + 1e-5f);

    float4 gg = ((const float4*)g)[tid];
    float4 bb = ((const float4*)be)[tid];
    float4 o;
    o.x = (v.x - mu) * rs * gg.x + bb.x;
    o.y = (v.y - mu) * rs * gg.y + bb.y;
    o.z = (v.z - mu) * rs * gg.z + bb.z;
    o.w = (v.w - mu) * rs * gg.w + bb.w;
    ((float4*)(xo + (size_t)row * CC))[tid] = o;
    _Float16* xhh = xh + (size_t)row * CC + tid * 4;
    xhh[0] = (_Float16)o.x; xhh[1] = (_Float16)o.y;
    xhh[2] = (_Float16)o.z; xhh[3] = (_Float16)o.w;
}

// ---------------------------------------------------------------------------
// Weight prep: f32 [K][N] -> f16 transposed [N][K]
// ---------------------------------------------------------------------------
__global__ __launch_bounds__(256) void cvtT_kernel(
    const float* __restrict__ src, _Float16* __restrict__ dst, int K, int N)
{
    size_t idx = (size_t)blockIdx.x * 256 + threadIdx.x;
    if (idx >= (size_t)K * N) return;
    int n = (int)(idx % N);
    int k = (int)(idx / N);
    dst[(size_t)n * K + k] = (_Float16)src[idx];
}

// Wq/Wk/Wv [H][C][HS] -> concatenated transposed f16 [3C][C],  j = w*C+h*64+d
__global__ __launch_bounds__(256) void qkvw_kernel(
    const float* __restrict__ Wq, const float* __restrict__ Wk,
    const float* __restrict__ Wv, _Float16* __restrict__ WT)
{
    size_t idx = (size_t)blockIdx.x * 256 + threadIdx.x;   // [0, 3*C*C)
    if (idx >= (size_t)3 * CC * CC) return;
    int c = (int)(idx % CC);
    size_t j = idx / CC;
    int w  = (int)(j / CC);
    int jj = (int)(j % CC);
    int h = jj >> 6, d = jj & 63;
    const float* W = (w == 0) ? Wq : ((w == 1) ? Wk : Wv);
    WT[idx] = (_Float16)W[((size_t)h * CC + c) * 64 + d];
}

// qkv f32 [M][3C] -> q_h[B,H,T,64] (pre-scaled by C^-0.5), k_h[B,H,T,64], v_t[B,H,64,T]
__global__ __launch_bounds__(256) void qkv_reshape_kernel(
    const float* __restrict__ qkv, _Float16* __restrict__ qh,
    _Float16* __restrict__ kh, _Float16* __restrict__ vt)
{
    size_t idx = (size_t)blockIdx.x * 256 + threadIdx.x;   // [0, M*3C)
    if (idx >= (size_t)BB * TT * 3 * CC) return;
    int j = (int)(idx % (3 * CC));
    size_t m = idx / (3 * CC);
    int b = (int)(m / TT), t = (int)(m % TT);
    int w = j >> 10;
    int jj = j & 1023;
    int h = jj >> 6, d = jj & 63;
    float val = qkv[idx];
    size_t bh = (size_t)(b * HH + h);
    if (w == 0)      qh[(bh * TT + t) * 64 + d] = (_Float16)(val * 0.03125f); // * C^-0.5
    else if (w == 1) kh[(bh * TT + t) * 64 + d] = (_Float16)val;
    else             vt[(bh * 64 + d) * TT + t] = (_Float16)val;
}

// ---------------------------------------------------------------------------
// WMMA GEMM with double-buffered async global->LDS staging.
//   out = act(A[M][K](f16) * BT[N][K]^T + bias (+resid))
// Block: 256 thr = 8 waves, tile 128x128, wave tile 64x32 (4x2 of 16x16).
// LDS tiles padded to stride 40 halves -> conflict-free row gathers
// (20*r mod 64 is a permutation of 16 distinct bank groups).
// ---------------------------------------------------------------------------
#define LDK 40   // padded LDS stride (halves) for a 32-wide k tile

__global__ __launch_bounds__(256) void gemm_kernel(
    const _Float16* __restrict__ A, const _Float16* __restrict__ BT,
    int M, int N, int K,
    const float* __restrict__ bias, const float* __restrict__ resid,
    float* __restrict__ outF, _Float16* __restrict__ outH, int relu)
{
    __shared__ _Float16 sA[2][128 * LDK];
    __shared__ _Float16 sB[2][128 * LDK];

    const int tid  = threadIdx.x;
    const int lane = tid & 31;
    const int wv   = tid >> 5;
    const int wm   = wv >> 2;          // 0..1
    const int wn   = wv & 3;           // 0..3
    const int bM   = blockIdx.y * 128;
    const int bN   = blockIdx.x * 128;
    const int lm   = lane & 15;
    const int akb  = (lane < 16) ? 0 : 8;    // A K sub-offset per lane-half
    const int bkb  = (lane < 16) ? 0 : 8;    // B K sub-offset per lane-half
    const int rowA0 = bM + wm * 64;
    const int colB0 = bN + wn * 32;

    // Staging assignment: each thread owns one 16-half chunk of the A tile
    // and one of the B tile (128 rows x 2 chunks = 256 chunks each).
    const int srow = tid >> 1;
    const int shc  = tid & 1;
    const _Float16* gA = A  + (size_t)(bM + srow) * K + shc * 16;
    const _Float16* gB = BT + (size_t)(bN + srow) * K + shc * 16;
    unsigned ldsA[2], ldsB[2];
#pragma unroll
    for (int bf = 0; bf < 2; bf++) {
        ldsA[bf] = (unsigned)(uintptr_t)&sA[bf][srow * LDK + shc * 16];
        ldsB[bf] = (unsigned)(uintptr_t)&sB[bf][srow * LDK + shc * 16];
    }

    v8f acc[4][2] = {};
    const int nk = K >> 5;

    // Prologue: stage k-step 0
    async_ld_b128(ldsA[0], gA);
    async_ld_b128(ldsB[0], gB);

    for (int it = 0; it < nk; ++it) {
        const int cur = it & 1;
        if (it + 1 < nk) {
            // Kick off next tile's DMA, then wait for current tile only
            // (async loads complete in order: <=2 outstanding => batch done).
            async_ld_b128(ldsA[cur ^ 1], gA + (size_t)(it + 1) * 32);
            async_ld_b128(ldsB[cur ^ 1], gB + (size_t)(it + 1) * 32);
            WAIT_ASYNC_LE_2();
        } else {
            WAIT_ASYNC_0();
        }
        __syncthreads();   // all waves' slices visible in LDS

        const _Float16* tA = &sA[cur][0];
        const _Float16* tB = &sB[cur][0];
        H16 a[4], b[2];
#pragma unroll
        for (int mt = 0; mt < 4; mt++) {
            const _Float16* pr = tA + (wm * 64 + mt * 16 + lm) * LDK + akb;
            a[mt].h[0] = *(const h8v*)(pr);
            a[mt].h[1] = *(const h8v*)(pr + 16);
        }
#pragma unroll
        for (int nt = 0; nt < 2; nt++) {
            const _Float16* pr = tB + (wn * 32 + nt * 16 + lm) * LDK + bkb;
            b[nt].h[0] = *(const h8v*)(pr);
            b[nt].h[1] = *(const h8v*)(pr + 16);
        }
#pragma unroll
        for (int mt = 0; mt < 4; mt++)
#pragma unroll
            for (int nt = 0; nt < 2; nt++)
                acc[mt][nt] = wmma_f16(a[mt].v, b[nt].v, acc[mt][nt]);

        __syncthreads();   // done reading; buffer may be overwritten next round
    }

    const int rbase = (lane < 16) ? 0 : 8;
#pragma unroll
    for (int mt = 0; mt < 4; mt++) {
#pragma unroll
        for (int nt = 0; nt < 2; nt++) {
            const int gcol = colB0 + nt * 16 + lm;
            const float bs = bias ? bias[gcol] : 0.f;
#pragma unroll
            for (int r = 0; r < 8; r++) {
                const int grow = rowA0 + mt * 16 + rbase + r;
                float v = acc[mt][nt][r] + bs;
                if (resid) v += resid[(size_t)grow * N + gcol];
                if (relu)  v = fmaxf(v, 0.f);
                if (outF)  outF[(size_t)grow * N + gcol] = v;
                if (outH)  outH[(size_t)grow * N + gcol] = (_Float16)v;
            }
        }
    }
}

// ---------------------------------------------------------------------------
// Flash attention (causal): one wave per 16 query rows of one (b,h).
// q_h pre-scaled by C^-0.5.  attn output written as f16 [M][C] (head-concat).
// ---------------------------------------------------------------------------
__global__ __launch_bounds__(256) void attn_kernel(
    const _Float16* __restrict__ qh, const _Float16* __restrict__ kh,
    const _Float16* __restrict__ vt, _Float16* __restrict__ attn)
{
    __shared__ _Float16 lds_p[8 * 16 * 32];   // 1 KB per wave
    const int tid  = threadIdx.x;
    const int lane = tid & 31;
    const int wv   = tid >> 5;
    const int qt   = blockIdx.x * 8 + wv;     // global q-tile id
    const int tile = qt & 127;
    const int h    = (qt >> 7) & 15;
    const int b    = qt >> 11;
    const int t0   = tile * 16;
    const size_t bh = (size_t)(b * HH + h);

    const _Float16* Q  = qh + bh * TT * 64;
    const _Float16* Km = kh + bh * TT * 64;
    const _Float16* Vt = vt + bh * 64 * TT;

    const int lm     = lane & 15;
    const int half16 = (lane < 16) ? 0 : 1;
    const int akb    = half16 * 8;
    const int bko    = half16 * 8;

    // Q fragments (16x64 -> two 16x32 A-frags), loaded once
    H16 qa[2];
    {
        const _Float16* qrow = Q + (size_t)(t0 + lm) * 64 + akb;
        qa[0].h[0] = *(const h8v*)(qrow);
        qa[0].h[1] = *(const h8v*)(qrow + 16);
        qa[1].h[0] = *(const h8v*)(qrow + 32);
        qa[1].h[1] = *(const h8v*)(qrow + 48);
    }

    v8f o[4] = {};
    float mrow[8], lrow[8];
#pragma unroll
    for (int r = 0; r < 8; r++) { mrow[r] = -1e30f; lrow[r] = 0.f; }

    _Float16* lp = lds_p + wv * 512;
    const int nch = (t0 + 16 + 31) >> 5;

    for (int ch = 0; ch < nch; ch++) {
        const int kk = ch * 32;

        // ---- S = Q K^T for 32 keys (two 16-col n-tiles) ----
        v8f s[2];
#pragma unroll
        for (int nt = 0; nt < 2; nt++) {
            const _Float16* krow = Km + (size_t)(kk + nt * 16 + lm) * 64 + bko;
            H16 b0, b1;
            b0.h[0] = *(const h8v*)(krow);      b0.h[1] = *(const h8v*)(krow + 16);
            b1.h[0] = *(const h8v*)(krow + 32); b1.h[1] = *(const h8v*)(krow + 48);
            v8f z = {};
            z = wmma_f16(qa[0].v, b0.v, z);
            z = wmma_f16(qa[1].v, b1.v, z);
            s[nt] = z;
        }

        // ---- causal mask (only diagonal chunks need it) ----
        if (kk + 31 > t0) {
#pragma unroll
            for (int nt = 0; nt < 2; nt++) {
                const int key = kk + nt * 16 + lm;
#pragma unroll
                for (int r = 0; r < 8; r++) {
                    const int qrow = t0 + r + half16 * 8;
                    if (key > qrow) s[nt][r] = -1e30f;
                }
            }
        }

        // ---- online softmax (row stats via half-wave xor reductions) ----
#pragma unroll
        for (int r = 0; r < 8; r++) {
            float tm = fmaxf(s[0][r], s[1][r]);
            for (int mm = 1; mm < 16; mm <<= 1) tm = fmaxf(tm, __shfl_xor(tm, mm));
            const float mnew = fmaxf(mrow[r], tm);
            const float f = __expf(mrow[r] - mnew);
            mrow[r] = mnew;
            const float p0 = __expf(s[0][r] - mnew);
            const float p1 = __expf(s[1][r] - mnew);
            s[0][r] = p0; s[1][r] = p1;
            float ls = p0 + p1;
            for (int mm = 1; mm < 16; mm <<= 1) ls += __shfl_xor(ls, mm);
            lrow[r] = lrow[r] * f + ls;
            o[0][r] *= f; o[1][r] *= f; o[2][r] *= f; o[3][r] *= f;
        }

        // ---- P (D-frag layout) -> LDS -> A-frag layout ----
#pragma unroll
        for (int nt = 0; nt < 2; nt++)
#pragma unroll
            for (int r = 0; r < 8; r++)
                lp[(r + half16 * 8) * 32 + nt * 16 + lm] = (_Float16)s[nt][r];
        asm volatile("s_wait_dscnt 0" ::: "memory");
        H16 pa;
        {
            const _Float16* pr = lp + lm * 32 + akb;
            pa.h[0] = *(const h8v*)pr;
            pa.h[1] = *(const h8v*)(pr + 16);
        }
        asm volatile("" ::: "memory");  // keep loads before next-iter stores

        // ---- O += P V  (V transposed: [64][T]) ----
#pragma unroll
        for (int dt = 0; dt < 4; dt++) {
            const _Float16* vr = Vt + (size_t)(dt * 16 + lm) * TT + kk + bko;
            H16 bb;
            bb.h[0] = *(const h8v*)vr;
            bb.h[1] = *(const h8v*)(vr + 16);
            o[dt] = wmma_f16(pa.v, bb.v, o[dt]);
        }
    }

    // ---- normalize and write head-concatenated f16 output ----
    float inv[8];
#pragma unroll
    for (int r = 0; r < 8; r++) inv[r] = 1.0f / lrow[r];
#pragma unroll
    for (int dt = 0; dt < 4; dt++) {
        const int d = dt * 16 + lm;
#pragma unroll
        for (int r = 0; r < 8; r++) {
            const int t = t0 + r + half16 * 8;
            attn[((size_t)(b * TT + t)) * CC + h * 64 + d] = (_Float16)(o[dt][r] * inv[r]);
        }
    }
}

// ---------------------------------------------------------------------------
// Launch
// ---------------------------------------------------------------------------
extern "C" void kernel_launch(void* const* d_in, const int* in_sizes, int n_in,
                              void* d_out, int out_size, void* d_ws, size_t ws_size,
                              hipStream_t stream) {
    (void)in_sizes; (void)n_in; (void)out_size; (void)ws_size;
    const float* x   = (const float*)d_in[0];
    const float* Wq  = (const float*)d_in[1];
    const float* Wk  = (const float*)d_in[2];
    const float* Wv  = (const float*)d_in[3];
    const float* Wo  = (const float*)d_in[4];
    const float* bo  = (const float*)d_in[5];
    const float* W1  = (const float*)d_in[6];
    const float* b1  = (const float*)d_in[7];
    const float* W2  = (const float*)d_in[8];
    const float* b2  = (const float*)d_in[9];
    const float* g1  = (const float*)d_in[10];
    const float* be1 = (const float*)d_in[11];
    const float* g2  = (const float*)d_in[12];
    const float* be2 = (const float*)d_in[13];

    const int M = BB * TT;            // 4096
    char* p = (char*)d_ws;
    auto take = [&](size_t bytes) -> void* {
        void* r = (void*)p;
        p += (bytes + 255) & ~(size_t)255;
        return r;
    };
    float*     xn1_f  = (float*)    take((size_t)M * CC * 4);
    _Float16*  xn1_h  = (_Float16*) take((size_t)M * CC * 2);
    _Float16*  WqkvT  = (_Float16*) take((size_t)3 * CC * CC * 2);
    _Float16*  WoT    = (_Float16*) take((size_t)CC * CC * 2);
    _Float16*  W1T    = (_Float16*) take((size_t)CC * 4 * CC * 2);
    _Float16*  W2T    = (_Float16*) take((size_t)4 * CC * CC * 2);
    float*     qkv_f  = (float*)    take((size_t)M * 3 * CC * 4);
    _Float16*  qh     = (_Float16*) take((size_t)M * CC * 2);
    _Float16*  khb    = (_Float16*) take((size_t)M * CC * 2);
    _Float16*  vtb    = (_Float16*) take((size_t)M * CC * 2);
    _Float16*  attn_h = (_Float16*) take((size_t)M * CC * 2);
    float*     x1_f   = (float*)    take((size_t)M * CC * 4);
    float*     xn2_f  = (float*)    take((size_t)M * CC * 4);
    _Float16*  xn2_h  = (_Float16*) take((size_t)M * CC * 2);
    _Float16*  h_h    = (_Float16*) take((size_t)M * 4 * CC * 2);

    // 1. LN1
    ln_kernel<<<M, 256, 0, stream>>>(x, g1, be1, xn1_f, xn1_h);
    // 2. weight prep (f16, transposed)
    qkvw_kernel<<<(3 * CC * CC + 255) / 256, 256, 0, stream>>>(Wq, Wk, Wv, WqkvT);
    cvtT_kernel<<<(CC * CC + 255) / 256, 256, 0, stream>>>(Wo, WoT, CC, CC);
    cvtT_kernel<<<(CC * 4 * CC + 255) / 256, 256, 0, stream>>>(W1, W1T, CC, 4 * CC);
    cvtT_kernel<<<(4 * CC * CC + 255) / 256, 256, 0, stream>>>(W2, W2T, 4 * CC, CC);
    // 3. fused QKV projection: [M,C] x [C,3C]
    gemm_kernel<<<dim3(3 * CC / 128, M / 128), 256, 0, stream>>>(
        xn1_h, WqkvT, M, 3 * CC, CC, nullptr, nullptr, qkv_f, nullptr, 0);
    // 4. reshape to per-head layouts (q pre-scaled by C^-0.5, v transposed)
    qkv_reshape_kernel<<<((size_t)M * 3 * CC + 255) / 256, 256, 0, stream>>>(
        qkv_f, qh, khb, vtb);
    // 5. causal flash attention -> attn_h f16 [M][C]
    attn_kernel<<<(BB * HH * (TT / 16)) / 8, 256, 0, stream>>>(qh, khb, vtb, attn_h);
    // 6. output projection + residual (x1 = xn1 + attn @ Wo + bo)
    gemm_kernel<<<dim3(CC / 128, M / 128), 256, 0, stream>>>(
        attn_h, WoT, M, CC, CC, bo, xn1_f, x1_f, nullptr, 0);
    // 7. LN2
    ln_kernel<<<M, 256, 0, stream>>>(x1_f, g2, be2, xn2_f, xn2_h);
    // 8. MLP up + ReLU (f16 out only)
    gemm_kernel<<<dim3(4 * CC / 128, M / 128), 256, 0, stream>>>(
        xn2_h, W1T, M, 4 * CC, CC, b1, nullptr, nullptr, h_h, 1);
    // 9. MLP down + residual -> d_out (f32)
    gemm_kernel<<<dim3(CC / 128, M / 128), 256, 0, stream>>>(
        h_h, W2T, M, CC, 4 * CC, b2, xn2_f, (float*)d_out, nullptr, 0);
}
